// AdativeFourierNeuralOperator_9268539425437
// MI455X (gfx1250) — matrix operand order repaired and checked by reading, not compiled, not run on hardware
//
#include <hip/hip_runtime.h>

// ---------------------------------------------------------------------------
// AFNO layer for MI455X (gfx1250, wave32, WMMA).
// All five stages (rfft-W, fft-H, block MLP x2 layers, ifft-H, irfft-W) are
// bf16 WMMA GEMMs (v_wmma_f32_16x16x32_bf16, fp32 accumulate).
// DFT matrices / weights pre-packed into WMMA fragment-native layouts.
// Round-3 changes: vectorized staging loads (float4 / 4x bf16 per thread)
// in the transpose-staging kernels; async global->LDS staging retained in
// the MLP kernel (GLOBAL_LOAD_ASYNC_TO_LDS_B128 + s_wait_asynccnt).
// ---------------------------------------------------------------------------

typedef __attribute__((ext_vector_type(16))) __bf16 v16bf;
typedef __attribute__((ext_vector_type(8)))  __bf16 v8bf;
typedef __attribute__((ext_vector_type(4)))  __bf16 v4bf;
typedef __attribute__((ext_vector_type(8)))  float  v8f;

#define LANE (threadIdx.x & 31)
#define WAVE (threadIdx.x >> 5)

static constexpr int   Bn = 32, Hh = 56, Ww = 56, Cc = 768, Wf = 29, BS = 192, NB = 4;
static constexpr int   FP = 32;                           // padded frequency dim (stage-1 out)
static constexpr float INV_SQ56 = 0.13363062095621219f;   // 1/sqrt(56) ('ortho')
static constexpr float LAMBDA   = 0.01f;                  // softshrink

__device__ __forceinline__ v8f wmma_bf16(v16bf a, v16bf b, v8f c) {
  return __builtin_amdgcn_wmma_f32_16x16x32_bf16(false, a, false, b, (short)0, c, false, false);
}

// Packed fragment (32 lanes x 16 bf16) load: one 32B global load per lane.
__device__ __forceinline__ v16bf load_frag(const __bf16* __restrict__ pack, int fragIdx) {
  return *(const v16bf*)(pack + ((long)fragIdx * 32 + LANE) * 16);
}

// A-operand gather from LDS tile stored [m][k]: two contiguous 16B ds loads.
__device__ __forceinline__ v16bf gather_A(const __bf16* __restrict__ lds, int stride, int kc) {
  int m  = LANE & 15;
  int kb = kc * 32 + ((LANE >= 16) ? 16 : 0);
  union { v16bf v; v8bf h[2]; } a;
  a.h[0] = *(const v8bf*)(lds + m * stride + kb);
  a.h[1] = *(const v8bf*)(lds + m * stride + kb + 8);
  return a.v;
}

// B-operand gather from LDS tile stored [n][k]: one 32B ds load.
__device__ __forceinline__ v16bf gather_B(const __bf16* __restrict__ lds, int stride, int kc) {
  int n  = LANE & 15;
  int kb = kc * 32 + ((LANE < 16) ? 0 : 16);
  return *(const v16bf*)(lds + n * stride + kb);
}

// CDNA5 async copy: 16B global -> LDS, tracked by ASYNCcnt (ISA 10.x / 15.18).
__device__ __forceinline__ void async_copy_b128(unsigned lds_addr, const void* gptr) {
  asm volatile("global_load_async_to_lds_b128 %0, %1, off"
               :: "v"(lds_addr), "v"((unsigned long long)(__SIZE_TYPE__)gptr)
               : "memory");
}
__device__ __forceinline__ void wait_async0() {
  asm volatile("s_wait_asynccnt 0x0" ::: "memory");
}

// Map a linear element index of an A-pack ([mtile][kc][lane][16]) to (m,k).
__device__ __forceinline__ void decode_frag(int local, int kcN, int& m, int& k) {
  int frag  = local >> 9;
  int lane  = (local >> 4) & 31;
  int e     = local & 15;
  int mtile = frag / kcN;
  int kc    = frag % kcN;
  m = mtile * 16 + (lane & 15);
  k = kc * 32 + ((lane >= 16) ? 16 : 0) + e;
}

// ------------------------- prep: DFT fragment packs -------------------------
__global__ void prep_dft(__bf16* s1c, __bf16* s1sn,
                         __bf16* hc, __bf16* hs, __bf16* hsn,
                         __bf16* pc, __bf16* ps) {
  int idx = blockIdx.x * 256 + threadIdx.x;
  if (idx >= 20480) return;
  const float TWO_PI = 6.283185307179586f;
  if (idx < 4096) {                       // rfft-W A packs: M=32 (f), K=64 (w), kcN=2
    int arr = idx >> 11; int local = idx & 2047;
    int m, k; decode_frag(local, 2, m, k);
    float v = 0.f;
    if (m < Wf && k < Ww) {
      float ang = TWO_PI * (float)((m * k) % Ww) / (float)Ww;
      v = (arr == 0 ? __cosf(ang) : -__sinf(ang)) * INV_SQ56;
    }
    (arr == 0 ? s1c : s1sn)[local] = (__bf16)v;
  } else if (idx < 16384) {               // fft-H A packs: M=64 (u), K=64 (h), kcN=2
    int rel = idx - 4096;
    int arr = rel / 4096; int local = rel % 4096;
    int m, k; decode_frag(local, 2, m, k);
    float v = 0.f;
    if (m < Hh && k < Hh) {
      float ang = TWO_PI * (float)((m * k) % Hh) / (float)Hh;
      float c = __cosf(ang), s = __sinf(ang);
      v = (arr == 0 ? c : (arr == 1 ? s : -s)) * INV_SQ56;
    }
    (arr == 0 ? hc : (arr == 1 ? hs : hsn))[local] = (__bf16)v;
  } else {                                // irfft-W A packs: M=64 (w), K=32 (f), kcN=1
    int rel = idx - 16384;
    int arr = rel >> 11; int local = rel & 2047;
    int m, k; decode_frag(local, 1, m, k);
    float v = 0.f;
    if (m < Ww && k < Wf) {
      float alpha = (k == 0 || k == Wf - 1) ? 1.f : 2.f;   // hermitian doubling
      float ang = TWO_PI * (float)((m * k) % Ww) / (float)Ww;
      v = (arr == 0 ? __cosf(ang) : -__sinf(ang)) * alpha * INV_SQ56;
    }
    (arr == 0 ? pc : ps)[local] = (__bf16)v;
  }
}

// ---------------------- prep: weight B-fragment packs -----------------------
__global__ void prep_wpack(const float* __restrict__ w,
                           __bf16* __restrict__ pr, __bf16* __restrict__ pi,
                           __bf16* __restrict__ pin) {
  int idx = blockIdx.x * 256 + threadIdx.x;
  if (idx >= 4 * 6 * 12 * 512) return;
  int e    = idx & 15;
  int lane = (idx >> 4) & 31;
  int nt   = (idx >> 9) % 12;
  int kc   = (idx / (512 * 12)) % 6;
  int blk  = idx / (512 * 12 * 6);
  int k = kc * 32 + ((lane < 16) ? 0 : 16) + e;
  int n = nt * 16 + (lane & 15);
  float vr = w[((0 * NB + blk) * BS + k) * BS + n];
  float vi = w[((1 * NB + blk) * BS + k) * BS + n];
  pr[idx]  = (__bf16)vr;
  pi[idx]  = (__bf16)vi;
  pin[idx] = (__bf16)(-vi);
}

// ---------------- stage 1: rfft along W as GEMM (real -> complex) -----------
__global__ void stage1_rfft_w(const float* __restrict__ x,
                              __bf16* __restrict__ y1r, __bf16* __restrict__ y1i,
                              const __bf16* __restrict__ AC, const __bf16* __restrict__ ASN) {
  __shared__ alignas(32) __bf16 lx[16 * 64];          // [c][w], w padded to 64
  int bh = blockIdx.x; int b = bh / Hh; int h = bh % Hh;
  int c0 = blockIdx.y * 16;
  long rowbase = (long)(b * Hh * Ww + h * Ww);
  for (int i = 0; i < 2; ++i) {
    int id = i * 128 + threadIdx.x;                   // 256 items: 4 cgroups x 64 w
    int cg = id & 3, w = id >> 2;
    float4 v = make_float4(0.f, 0.f, 0.f, 0.f);
    if (w < Ww) v = *(const float4*)&x[(rowbase + w) * Cc + c0 + cg * 4];
    lx[(cg * 4 + 0) * 64 + w] = (__bf16)v.x;
    lx[(cg * 4 + 1) * 64 + w] = (__bf16)v.y;
    lx[(cg * 4 + 2) * 64 + w] = (__bf16)v.z;
    lx[(cg * 4 + 3) * 64 + w] = (__bf16)v.w;
  }
  __syncthreads();
  int part = WAVE & 1, mtile = WAVE >> 1;
  const __bf16* A = part ? ASN : AC;
  v8f acc = {};
  for (int kc = 0; kc < 2; ++kc)
    acc = wmma_bf16(load_frag(A, mtile * 2 + kc), gather_B(lx, 64, kc), acc);
  union { v8f v; float f[8]; } r; r.v = acc;
  int n = LANE & 15, mo = (LANE >= 16) ? 8 : 0;
  __bf16* dst = part ? y1i : y1r;
  long obase = ((long)(b * FP + mtile * 16 + mo) * Hh + h) * Cc + c0 + n;
  for (int rr = 0; rr < 8; ++rr)                      // layout [b][f:32][h][c]
    dst[obase + (long)rr * Hh * Cc] = (__bf16)r.f[rr];
}

// -------- stage 2/4: complex GEMM along H (used fwd and inv via A ptrs) -----
__global__ void stage_cfft_h(const __bf16* __restrict__ br_p, const __bf16* __restrict__ bi_p,
                             __bf16* __restrict__ dr_p, __bf16* __restrict__ di_p,
                             const __bf16* __restrict__ Arr, const __bf16* __restrict__ Ari,
                             const __bf16* __restrict__ Air, const __bf16* __restrict__ Aii,
                             int sb, int sf, int sk) {
  __shared__ alignas(32) __bf16 lb[2 * 16 * 64];       // [re/im][c][h], h padded to 64
  int bf = blockIdx.x; int b = bf / Wf; int f = bf % Wf;
  int c0 = blockIdx.y * 16;
  long inbase = (long)b * sb + (long)f * sf + c0;
  for (int i = 0; i < 2; ++i) {
    int id = i * 256 + threadIdx.x;                    // 512 items: 2 planes x 4 cg x 64 u
    int pl = id >> 8; int rem = id & 255;
    int cg = rem & 3, u = rem >> 2;
    const __bf16* src = pl ? bi_p : br_p;
    union { v4bf v; __bf16 e[4]; } q; q.v = (v4bf){};
    if (u < Hh) q.v = *(const v4bf*)&src[inbase + (long)u * sk + cg * 4];
    __bf16* L = lb + pl * 1024 + (cg * 4) * 64 + u;
    L[0]   = q.e[0];
    L[64]  = q.e[1];
    L[128] = q.e[2];
    L[192] = q.e[3];
  }
  __syncthreads();
  int part = WAVE & 1, mtile = WAVE >> 1;
  const __bf16* A0 = part ? Air : Arr;
  const __bf16* A1 = part ? Aii : Ari;
  v8f acc = {};
  for (int kc = 0; kc < 2; ++kc) {
    int fi = mtile * 2 + kc;
    acc = wmma_bf16(load_frag(A0, fi), gather_B(lb,        64, kc), acc);
    acc = wmma_bf16(load_frag(A1, fi), gather_B(lb + 1024, 64, kc), acc);
  }
  union { v8f v; float fv[8]; } r; r.v = acc;
  int n = LANE & 15, mo = (LANE >= 16) ? 8 : 0;
  __bf16* dst = part ? di_p : dr_p;
  if (mtile < 3 || LANE < 16) {                        // uniform-per-half guard
    long obase = ((long)(b * Hh + mtile * 16 + mo) * Wf + f) * Cc + c0 + n;
    for (int rr = 0; rr < 8; ++rr)
      dst[obase + (long)rr * Wf * Cc] = (__bf16)r.fv[rr];
  }
}

// ----------------- stage 3: fused 2-layer complex block MLP -----------------
__global__ void mlp_block(const __bf16* __restrict__ xr_p, const __bf16* __restrict__ xi_p,
                          __bf16* __restrict__ zr_p, __bf16* __restrict__ zi_p,
                          const __bf16* __restrict__ W1r, const __bf16* __restrict__ W1i,
                          const __bf16* __restrict__ W1iN,
                          const __bf16* __restrict__ W2r, const __bf16* __restrict__ W2i,
                          const __bf16* __restrict__ W2iN,
                          const float* __restrict__ b1, const float* __restrict__ b2) {
  __shared__ alignas(32) __bf16 axr[16 * BS];
  __shared__ alignas(32) __bf16 axi[16 * BS];
  __shared__ alignas(32) __bf16 ahr[16 * BS];
  __shared__ alignas(32) __bf16 ahi[16 * BS];
  int rt = blockIdx.x, blk = blockIdx.y;
  long base = (long)rt * 16 * Cc + blk * BS;
  // ---- async staging: 2 planes x 16 rows x 24 16B-chunks = 768 copies
  {
    unsigned lr = (unsigned)(__SIZE_TYPE__)(void*)axr;
    unsigned li = (unsigned)(__SIZE_TYPE__)(void*)axi;
    for (int i = 0; i < 3; ++i) {
      int id  = i * 256 + threadIdx.x;                 // 0..767
      int pl  = id >= 384;
      int rem = pl ? id - 384 : id;
      int m   = rem / 24, kch = rem % 24;
      const __bf16* src = pl ? xi_p : xr_p;
      unsigned lbase    = pl ? li : lr;
      async_copy_b128(lbase + (unsigned)(m * (BS * 2) + kch * 16),
                      (const char*)(src + base + (long)m * Cc) + kch * 16);
    }
    wait_async0();
  }
  __syncthreads();
  int n = LANE & 15, mo = (LANE >= 16) ? 8 : 0;
  // ---- layer 1: hr = relu(xr*W1r - xi*W1i + b1r); hi = relu(xr*W1i + xi*W1r + b1i)
  for (int j = 0; j < 3; ++j) {
    int job = j * 8 + WAVE;                            // 24 jobs = 12 ntiles x {re,im}
    int nt = job % 12, part = job / 12;
    const __bf16* Wa = part ? W1i : W1r;
    const __bf16* Wb = part ? W1r : W1iN;
    v8f acc = {};
    for (int kc = 0; kc < 6; ++kc) {
      int fi = (blk * 6 + kc) * 12 + nt;
      if (kc + 1 < 6) {
        __builtin_prefetch(Wa + (long)(fi + 12) * 512, 0, 1);
        __builtin_prefetch(Wb + (long)(fi + 12) * 512, 0, 1);
      }
      acc = wmma_bf16(gather_A(axr, BS, kc), load_frag(Wa, fi), acc);
      acc = wmma_bf16(gather_A(axi, BS, kc), load_frag(Wb, fi), acc);
    }
    float bias = b1[(part * NB + blk) * BS + nt * 16 + n];
    union { v8f v; float f[8]; } r; r.v = acc;
    __bf16* hd = part ? ahi : ahr;
    for (int rr = 0; rr < 8; ++rr) {
      float v = r.f[rr] + bias;
      v = v > 0.f ? v : 0.f;                           // relu
      hd[(rr + mo) * BS + nt * 16 + n] = (__bf16)v;
    }
  }
  __syncthreads();
  // ---- layer 2 + softshrink
  for (int j = 0; j < 3; ++j) {
    int job = j * 8 + WAVE;
    int nt = job % 12, part = job / 12;
    const __bf16* Wa = part ? W2i : W2r;
    const __bf16* Wb = part ? W2r : W2iN;
    v8f acc = {};
    for (int kc = 0; kc < 6; ++kc) {
      int fi = (blk * 6 + kc) * 12 + nt;
      if (kc + 1 < 6) {
        __builtin_prefetch(Wa + (long)(fi + 12) * 512, 0, 1);
        __builtin_prefetch(Wb + (long)(fi + 12) * 512, 0, 1);
      }
      acc = wmma_bf16(gather_A(ahr, BS, kc), load_frag(Wa, fi), acc);
      acc = wmma_bf16(gather_A(ahi, BS, kc), load_frag(Wb, fi), acc);
    }
    float bias = b2[(part * NB + blk) * BS + nt * 16 + n];
    union { v8f v; float f[8]; } r; r.v = acc;
    __bf16* zd = part ? zi_p : zr_p;
    for (int rr = 0; rr < 8; ++rr) {
      float v = r.f[rr] + bias;
      float av = fabsf(v) - LAMBDA;                    // softshrink
      v = (av > 0.f) ? copysignf(av, v) : 0.f;
      zd[base + (long)(rr + mo) * Cc + nt * 16 + n] = (__bf16)v;
    }
  }
}

// -------------- stage 5: irfft along W (complex -> real, fp32 out) ----------
__global__ void stage5_irfft_w(const __bf16* __restrict__ zr_p, const __bf16* __restrict__ zi_p,
                               float* __restrict__ out,
                               const __bf16* __restrict__ PC, const __bf16* __restrict__ PS) {
  __shared__ alignas(32) __bf16 lz[2 * 16 * 32];       // [re/im][c][f], f padded to 32
  int bh = blockIdx.x; int b = bh / Hh; int h = bh % Hh;
  int c0 = blockIdx.y * 16;
  long inbase = (long)(b * Hh + h) * Wf * Cc + c0;
  for (int i = 0; i < 2; ++i) {
    int id = i * 128 + threadIdx.x;                    // 256 items: 2 planes x 4 cg x 32 f
    int pl = id >> 7; int rem = id & 127;
    int cg = rem & 3, fq = rem >> 2;
    const __bf16* src = pl ? zi_p : zr_p;
    union { v4bf v; __bf16 e[4]; } q; q.v = (v4bf){};
    if (fq < Wf) q.v = *(const v4bf*)&src[inbase + (long)fq * Cc + cg * 4];
    __bf16* L = lz + pl * 512 + (cg * 4) * 32 + fq;
    L[0]  = q.e[0];
    L[32] = q.e[1];
    L[64] = q.e[2];
    L[96] = q.e[3];
  }
  __syncthreads();
  int mtile = WAVE;                                    // 4 waves, w-tiles 0..3
  v8f acc = {};
  acc = wmma_bf16(load_frag(PC, mtile), gather_B(lz,       32, 0), acc);
  acc = wmma_bf16(load_frag(PS, mtile), gather_B(lz + 512, 32, 0), acc);
  union { v8f v; float f[8]; } r; r.v = acc;
  int n = LANE & 15, mo = (LANE >= 16) ? 8 : 0;
  if (mtile < 3 || LANE < 16) {                        // uniform-per-half guard
    long obase = ((long)(b * Hh + h) * Ww + mtile * 16 + mo) * Cc + c0 + n;
    for (int rr = 0; rr < 8; ++rr)
      out[obase + (long)rr * Cc] = r.f[rr];
  }
}

// ---------------------------------------------------------------------------
extern "C" void kernel_launch(void* const* d_in, const int* in_sizes, int n_in,
                              void* d_out, int out_size, void* d_ws, size_t ws_size,
                              hipStream_t stream) {
  (void)in_sizes; (void)n_in; (void)out_size; (void)ws_size;
  const float* x  = (const float*)d_in[0];
  const float* w1 = (const float*)d_in[1];
  const float* b1 = (const float*)d_in[2];
  const float* w2 = (const float*)d_in[3];
  const float* b2 = (const float*)d_in[4];
  float* out = (float*)d_out;

  char* ws = (char*)d_ws;
  auto alloc = [&](size_t bytes) {
    char* p = ws; ws += (bytes + 255) & ~(size_t)255; return (__bf16*)p;
  };
  __bf16* S1C  = alloc(2048 * 2);
  __bf16* S1SN = alloc(2048 * 2);
  __bf16* HC   = alloc(4096 * 2);
  __bf16* HS   = alloc(4096 * 2);
  __bf16* HSN  = alloc(4096 * 2);
  __bf16* PC   = alloc(2048 * 2);
  __bf16* PS   = alloc(2048 * 2);
  const size_t wp = (size_t)4 * 6 * 12 * 512;
  __bf16* pW1r = alloc(wp * 2); __bf16* pW1i = alloc(wp * 2); __bf16* pW1iN = alloc(wp * 2);
  __bf16* pW2r = alloc(wp * 2); __bf16* pW2i = alloc(wp * 2); __bf16* pW2iN = alloc(wp * 2);
  const size_t plane1 = (size_t)Bn * FP * Hh * Cc;     // stage-1 out, f padded to 32
  const size_t plane  = (size_t)Bn * Hh * Wf * Cc;     // packed [..][29][c] planes
  __bf16* Y1r = alloc(plane1 * 2); __bf16* Y1i = alloc(plane1 * 2);
  __bf16* Y2r = alloc(plane  * 2); __bf16* Y2i = alloc(plane  * 2);

  prep_dft<<<80, 256, 0, stream>>>(S1C, S1SN, HC, HS, HSN, PC, PS);
  prep_wpack<<<576, 256, 0, stream>>>(w1, pW1r, pW1i, pW1iN);
  prep_wpack<<<576, 256, 0, stream>>>(w2, pW2r, pW2i, pW2iN);

  // stage 1: x -> Y1 ([b][f:32][h][c])
  stage1_rfft_w<<<dim3(Bn * Hh, Cc / 16), 128, 0, stream>>>(x, Y1r, Y1i, S1C, S1SN);
  // stage 2 (fwd fft-H): D_r = C*Yr + S*Yi ; D_i = -S*Yr + C*Yi  -> Y2 ([b][u][f:29][c])
  stage_cfft_h<<<dim3(Bn * Wf, Cc / 16), 256, 0, stream>>>(
      Y1r, Y1i, Y2r, Y2i, HC, HS, HSN, HC,
      /*sb=*/FP * Hh * Cc, /*sf=*/Hh * Cc, /*sk=*/Cc);
  // stage 3: block MLP, Y2 -> Z (reuses Y1 planes, [b][u][f:29][c])
  mlp_block<<<dim3((Bn * Hh * Wf) / 16, NB), 256, 0, stream>>>(
      Y2r, Y2i, Y1r, Y1i, pW1r, pW1i, pW1iN, pW2r, pW2i, pW2iN, b1, b2);
  // stage 4 (inv fft-H): D_r = C*Zr - S*Zi ; D_i = S*Zr + C*Zi  -> Y2 ([b][h][f:29][c])
  stage_cfft_h<<<dim3(Bn * Wf, Cc / 16), 256, 0, stream>>>(
      Y1r, Y1i, Y2r, Y2i, HC, HSN, HS, HC,
      /*sb=*/Hh * Wf * Cc, /*sf=*/Cc, /*sk=*/Wf * Cc);
  // stage 5: irfft-W with hermitian fold -> fp32 out ([b][h][w][c])
  stage5_irfft_w<<<dim3(Bn * Hh, Cc / 16), 128, 0, stream>>>(Y2r, Y2i, out, PC, PS);
}
